// QuantizedLinear_10720238371576
// MI455X (gfx1250) — 8-bit matrix operand order repaired and checked by reading, compiled, not run on hardware
//
#include <hip/hip_runtime.h>

typedef __attribute__((ext_vector_type(8))) int v8i;

static constexpr int M = 8192;
static constexpr int K = 4096;
static constexpr int N = 4096;
static constexpr int BM = 128;
static constexpr int BN = 256;
static constexpr int BK = 64;
static constexpr int NK = K / BK;

__device__ __forceinline__ v8i wmma_iu8(v8i a, v8i b, v8i c) {
  return __builtin_amdgcn_wmma_i32_16x16x64_iu8(false, a, false, b, c, false, false);
}

__device__ __forceinline__ void async_ld_b128(void* lds_ptr, const void* gptr) {
  unsigned lds_addr = (unsigned)(size_t)lds_ptr;
  asm volatile("global_load_async_to_lds_b128 %0, %1, off"
               :
               : "v"(lds_addr), "v"(gptr)
               : "memory");
}

__device__ __forceinline__ void wait_async0() {
#if __has_builtin(__builtin_amdgcn_s_wait_asynccnt)
  __builtin_amdgcn_s_wait_asynccnt(0);
#else
  asm volatile("s_wait_asynccnt 0" ::: "memory");
#endif
}

__global__ __launch_bounds__(256) void pack_rows_256(const int* __restrict__ src,
                                                     unsigned char* __restrict__ dst,
                                                     int* __restrict__ rowsum) {
  __shared__ int red[256];
  const int row = blockIdx.x;
  const int4* s = (const int4*)(src + (size_t)row * K);
  unsigned int* d = (unsigned int*)(dst + (size_t)row * K);
  int sum = 0;
  for (int i = threadIdx.x; i < (K >> 2); i += 256) {
    int4 v = s[i];
    unsigned int b0 = (unsigned int)(v.x & 255);
    unsigned int b1 = (unsigned int)(v.y & 255);
    unsigned int b2 = (unsigned int)(v.z & 255);
    unsigned int b3 = (unsigned int)(v.w & 255);
    d[i] = b0 | (b1 << 8) | (b2 << 16) | (b3 << 24);
    sum += (int)(b0 + b1 + b2 + b3);
  }
  red[threadIdx.x] = sum;
  __syncthreads();
  for (int off = 128; off > 0; off >>= 1) {
    if ((int)threadIdx.x < off) red[threadIdx.x] += red[threadIdx.x + off];
    __syncthreads();
  }
  if (threadIdx.x == 0) rowsum[row] = red[0];
}

__global__ __launch_bounds__(256) void qgemm_iu8_wmma(
    const unsigned char* __restrict__ xp, const unsigned char* __restrict__ wp,
    const int* __restrict__ rsx, const int* __restrict__ rsw,
    const float* __restrict__ wscale, const int* __restrict__ wzp,
    const float* __restrict__ bias, const float* __restrict__ in_scale,
    const int* __restrict__ in_zp, float* __restrict__ out) {
  __shared__ uint4 sA[2][BM * 4];
  __shared__ uint4 sB[2][BN * 4];

  const int t = threadIdx.x;
  const int rowBase = blockIdx.x * BM;
  const int colBase = blockIdx.y * BN;

  const int lr = t >> 2;
  const int ls = t & 3;

  const int lane = t & 31;
  const int wave = t >> 5;
  const int wr = wave & 1;
  const int wc = wave >> 1;
  const int nlo = lane & 15;
  const int half = lane >> 4;

  v8i zero = {0, 0, 0, 0, 0, 0, 0, 0};
  v8i acc[4][4];
#pragma unroll
  for (int i = 0; i < 4; ++i)
#pragma unroll
    for (int j = 0; j < 4; ++j) acc[i][j] = zero;

  const unsigned char* gA0 = xp + (size_t)(rowBase + lr) * K;
  const unsigned char* gA1 = xp + (size_t)(rowBase + lr + 64) * K;
  const unsigned char* gB0 = wp + (size_t)(colBase + lr) * K;
  const unsigned char* gB1 = wp + (size_t)(colBase + lr + 64) * K;
  const unsigned char* gB2 = wp + (size_t)(colBase + lr + 128) * K;
  const unsigned char* gB3 = wp + (size_t)(colBase + lr + 192) * K;
  const int seg = ls * 16;

  auto stage = [&](int buf, int ko) {
    async_ld_b128(&sA[buf][lr * 4 + ls], gA0 + ko + seg);
    async_ld_b128(&sA[buf][(lr + 64) * 4 + ls], gA1 + ko + seg);
    async_ld_b128(&sB[buf][lr * 4 + ls], gB0 + ko + seg);
    async_ld_b128(&sB[buf][(lr + 64) * 4 + ls], gB1 + ko + seg);
    async_ld_b128(&sB[buf][(lr + 128) * 4 + ls], gB2 + ko + seg);
    async_ld_b128(&sB[buf][(lr + 192) * 4 + ls], gB3 + ko + seg);
  };

  stage(0, 0);
  wait_async0();
  __syncthreads();

  int cur = 0;
  for (int kt = 0; kt < NK; ++kt) {
    const bool hasNext = (kt + 1) < NK;
    if (hasNext) stage(cur ^ 1, (kt + 1) * BK);
    if (kt + 2 < NK) {
      const int kp = (kt + 2) * BK;
      __builtin_prefetch(gA0 + kp + seg, 0, 1);
      __builtin_prefetch(gB0 + kp + seg, 0, 1);
    }

    const unsigned char* bufA = (const unsigned char*)sA[cur];
    const unsigned char* bufB = (const unsigned char*)sB[cur];

    v8i bf[4];
#pragma unroll
    for (int j = 0; j < 4; ++j) {
      const unsigned char* rp = bufB + (size_t)(wc * 64 + j * 16 + nlo) * BK;
      int4 q0 = *(const int4*)(rp + half * 16);
      int4 q1 = *(const int4*)(rp + 32 + half * 16);
      v8i b = {q0.x, q0.y, q0.z, q0.w, q1.x, q1.y, q1.z, q1.w};
      bf[j] = b;
    }

#pragma unroll
    for (int i = 0; i < 4; ++i) {
      const unsigned char* rp =
          bufA + (size_t)(wr * 64 + i * 16 + nlo) * BK;
      int4 d0 = *(const int4*)(rp + half * 16);
      int4 d1 = *(const int4*)(rp + 32 + half * 16);
      v8i a = {d0.x, d0.y, d0.z, d0.w, d1.x, d1.y, d1.z, d1.w};
      acc[i][0] = wmma_iu8(a, bf[0], acc[i][0]);
      acc[i][1] = wmma_iu8(a, bf[1], acc[i][1]);
      acc[i][2] = wmma_iu8(a, bf[2], acc[i][2]);
      acc[i][3] = wmma_iu8(a, bf[3], acc[i][3]);
    }

    if (hasNext) wait_async0();
    __syncthreads();
    cur ^= 1;
  }

  const int izp = *in_zp;
  const float isc = *in_scale;
#pragma unroll
  for (int j = 0; j < 4; ++j) {
    const int col = colBase + wc * 64 + j * 16 + nlo;
    const int wz = wzp[col];
    const int cn = izp * (K * wz - rsw[col]);
    const float sc = isc * wscale[col];
    const float bs = bias[col];
#pragma unroll
    for (int i = 0; i < 4; ++i) {
      const int rowb = rowBase + wr * 64 + i * 16 + half * 8;
#pragma unroll
      for (int v = 0; v < 8; ++v) {
        const int row = rowb + v;
        const int ival = acc[i][j][v] + cn - wz * rsx[row];
        out[(size_t)row * N + col] = sc * (float)ival + bs;
      }
    }
  }
}

extern "C" void kernel_launch(void* const* d_in, const int* in_sizes, int n_in,
                              void* d_out, int out_size, void* d_ws, size_t ws_size,
                              hipStream_t stream) {
  const int* x_q = (const int*)d_in[0];
  const int* w_q = (const int*)d_in[1];
  const float* w_scale = (const float*)d_in[2];
  const int* w_zp = (const int*)d_in[3];
  const float* bias = (const float*)d_in[4];
  const float* in_scale = (const float*)d_in[5];
  const int* in_zp = (const int*)d_in[6];
  float* out = (float*)d_out;

  unsigned char* xp = (unsigned char*)d_ws;
  unsigned char* wp = xp + (size_t)M * K;
  int* rsx = (int*)(wp + (size_t)N * K);
  int* rsw = rsx + M;

  pack_rows_256<<<M, 256, 0, stream>>>(x_q, xp, rsx);
  pack_rows_256<<<N, 256, 0, stream>>>(w_q, wp, rsw);
  qgemm_iu8_wmma<<<dim3(M / BM, N / BN), 256, 0, stream>>>(
      xp, wp, rsx, rsw, w_scale, w_zp, bias, in_scale, in_zp, out);
}
